// GCN_59407987638457
// MI455X (gfx1250) — matrix operand order repaired and checked by reading, not compiled, not run on hardware
//
#include <hip/hip_runtime.h>

typedef __attribute__((ext_vector_type(16))) _Float16 v16h;
typedef __attribute__((ext_vector_type(8)))  float    v8f;

// ---------------------------------------------------------------------------
// WMMA fragment helpers (contiguous k order: k = 16*(lane/16) + e, wave32)
// ---------------------------------------------------------------------------

// A-matrix 16x32 f16 fragment from a row-major f16 tile in LDS.
// lane 0-15: M=lane, K in {0..15}; lane 16-31: M=lane-16, K in {16..31}
// Per lane this is two contiguous 16-byte runs -> ds_load_b128 pairs.
__device__ __forceinline__ v16h load_a_frag(const _Float16* __restrict__ base,
                                            int m, int halfsel, int ktile, int stride) {
  v16h a;
  const _Float16* row = base + m * stride + ktile * 32;
#pragma unroll
  for (int e = 0; e < 16; ++e) {
    const int v = e >> 1, off = e & 1;
    const int k = (v * 2 + off) + 16 * halfsel;
    a[e] = row[k];
  }
  return a;
}

// B-matrix 32x16 f16 fragment from a *fragment-swizzled* LDS tile:
// element (k,n) lives at ((kt*NT + nt)*32 + lane)*16 + e with
// lane = ((k>>4)&1)*16 + (n&15), e = k&15. One v16h = 2x ds_load_b128.
__device__ __forceinline__ v16h load_b_frag_swz(const _Float16* __restrict__ Ws,
                                                int kt, int nt, int lane, int ntiles) {
  return *(const v16h*)(Ws + (((kt * ntiles + nt) * 32) + lane) * 16);
}

// Stage a [128 x 128] row-major f32 weight into the swizzled f16 layout.
// Handles k in pairs so each LDS write is a contiguous 2xf16 (ds_store_b32).
__device__ __forceinline__ void stage_weight128_swz(const float* __restrict__ W,
                                                    _Float16* __restrict__ Ws,
                                                    int tid, int nthreads) {
  for (int i = tid; i < 64 * 128; i += nthreads) {
    const int n = i & 127;
    const int k = (i >> 7) << 1;  // even k
    const _Float16 v0 = (_Float16)W[k * 128 + n];
    const _Float16 v1 = (_Float16)W[(k + 1) * 128 + n];
    const int kt = k >> 5, kk = k & 31;
    const int hs = kk >> 4, e = kk & 15;      // e even; e+1 same lane
    const int ln = hs * 16 + (n & 15);
    const int nt = n >> 4;
    _Float16* dst = Ws + (((kt * 8 + nt) * 32) + ln) * 16 + e;
    dst[0] = v0;
    dst[1] = v1;
  }
}

// ---------------------------------------------------------------------------
// Kernel 1: init degrees (self-edge => 1.0) and zero aggregation buffer
// ---------------------------------------------------------------------------
__global__ void gcn_init_kernel(float* __restrict__ agg, float* __restrict__ sdeg,
                                float* __restrict__ rdeg, long Nn) {
  const long id = (long)blockIdx.x * blockDim.x + threadIdx.x;
  if (id < Nn * 128) agg[id] = 0.0f;
  if (id < Nn) { sdeg[id] = 1.0f; rdeg[id] = 1.0f; }
}

// ---------------------------------------------------------------------------
// Kernel 2: degree accumulation over the E real edges
// ---------------------------------------------------------------------------
__global__ void gcn_degree_kernel(const int* __restrict__ snd, const int* __restrict__ rcv,
                                  float* __restrict__ sdeg, float* __restrict__ rdeg, long E) {
  const long id = (long)blockIdx.x * blockDim.x + threadIdx.x;
  if (id < E) {
    atomicAdd(&sdeg[snd[id]], 1.0f);
    atomicAdd(&rdeg[rcv[id]], 1.0f);
  }
}

// ---------------------------------------------------------------------------
// Kernel 3: node MLP  h = (relu(X@W1+b1)@W2+b2) * rsqrt(max(sdeg,1))
// 128 threads = 4 waves; each wave owns 32 rows (two 16-row A tiles sharing
// every B fragment) of a 128-row block.
// ---------------------------------------------------------------------------
__global__ __launch_bounds__(128) void node_mlp_wmma(
    const float* __restrict__ X, const float* __restrict__ W1, const float* __restrict__ b1,
    const float* __restrict__ W2, const float* __restrict__ b2,
    const float* __restrict__ sdeg, float* __restrict__ Hout) {
  __shared__ _Float16 Xs[128 * 128];      // X block tile, later intermediate (32 KB)
  __shared__ _Float16 Ws[4 * 8 * 32 * 16];  // swizzled weight tile (32 KB)

  const int tid = threadIdx.x;
  const int wave = tid >> 5;          // 0..3
  const int lane = tid & 31;
  const int hs = lane >> 4;
  const int lo = lane & 15;
  const long row0 = (long)blockIdx.x * 128;

  // Prefetch next block's X while we work on this one (global_prefetch_b8).
  if (blockIdx.x + 1 < gridDim.x)
    __builtin_prefetch(X + (row0 + 128) * 128, 0, 0);

  // Stage X block (f32 -> f16, row-major) and W1 (swizzled).
  {
    const float4* Xv = (const float4*)(X + row0 * 128);
    for (int i = tid; i < 128 * 32; i += 128) {
      float4 x = Xv[i];
      Xs[i * 4 + 0] = (_Float16)x.x; Xs[i * 4 + 1] = (_Float16)x.y;
      Xs[i * 4 + 2] = (_Float16)x.z; Xs[i * 4 + 3] = (_Float16)x.w;
    }
    stage_weight128_swz(W1, Ws, tid, 128);
  }
  __syncthreads();

  // GEMM1: preload A fragments for this wave's 32 rows (registers), then it
  // is safe to overwrite our own LDS rows with the intermediate.
  v16h aAll[2][4];
#pragma unroll
  for (int mt = 0; mt < 2; ++mt)
#pragma unroll
    for (int kt = 0; kt < 4; ++kt)
      aAll[mt][kt] = load_a_frag(Xs + (wave * 32 + mt * 16) * 128, lo, hs, kt, 128);

#pragma unroll
  for (int nt = 0; nt < 8; ++nt) {
    v8f acc0 = {}, acc1 = {};
#pragma unroll
    for (int kt = 0; kt < 4; ++kt) {
      v16h b = load_b_frag_swz(Ws, kt, nt, lane, 8);
      acc0 = __builtin_amdgcn_wmma_f32_16x16x32_f16(false, aAll[0][kt], false, b,
                                                    (short)0, acc0, false, false);
      acc1 = __builtin_amdgcn_wmma_f32_16x16x32_f16(false, aAll[1][kt], false, b,
                                                    (short)0, acc1, false, false);
    }
    const int n = nt * 16 + lo;
    const float bias = b1[n];
#pragma unroll
    for (int r = 0; r < 8; ++r) {
      const int m = r + 8 * hs;
      float v0 = acc0[r] + bias;
      float v1 = acc1[r] + bias;
      v0 = v0 > 0.0f ? v0 : 0.0f;
      v1 = v1 > 0.0f ? v1 : 0.0f;
      Xs[(wave * 32 + m) * 128 + n]      = (_Float16)v0;  // own rows only
      Xs[(wave * 32 + 16 + m) * 128 + n] = (_Float16)v1;
    }
  }
  __syncthreads();  // all waves done reading the W1 tile

  stage_weight128_swz(W2, Ws, tid, 128);
  __syncthreads();

  // GEMM2, fused with sender-degree normalization at store.
#pragma unroll
  for (int mt = 0; mt < 2; ++mt)
#pragma unroll
    for (int kt = 0; kt < 4; ++kt)
      aAll[mt][kt] = load_a_frag(Xs + (wave * 32 + mt * 16) * 128, lo, hs, kt, 128);

#pragma unroll
  for (int nt = 0; nt < 8; ++nt) {
    v8f acc0 = {}, acc1 = {};
#pragma unroll
    for (int kt = 0; kt < 4; ++kt) {
      v16h b = load_b_frag_swz(Ws, kt, nt, lane, 8);
      acc0 = __builtin_amdgcn_wmma_f32_16x16x32_f16(false, aAll[0][kt], false, b,
                                                    (short)0, acc0, false, false);
      acc1 = __builtin_amdgcn_wmma_f32_16x16x32_f16(false, aAll[1][kt], false, b,
                                                    (short)0, acc1, false, false);
    }
    const int n = nt * 16 + lo;
    const float bias = b2[n];
#pragma unroll
    for (int r = 0; r < 8; ++r) {
      const int m = r + 8 * hs;
      const long g0 = row0 + wave * 32 + m;
      const long g1 = g0 + 16;
      const float s0 = rsqrtf(fmaxf(sdeg[g0], 1.0f));
      const float s1 = rsqrtf(fmaxf(sdeg[g1], 1.0f));
      Hout[g0 * 128 + n] = (acc0[r] + bias) * s0;
      Hout[g1 * 128 + n] = (acc1[r] + bias) * s1;
    }
  }
}

// ---------------------------------------------------------------------------
// Kernel 4: edge scatter-add  agg[rcv] += h[snd]  (plus N self edges)
// One thread handles 4 features of one edge: float4 gather + 4 float atomics.
// ---------------------------------------------------------------------------
__global__ void gcn_scatter_kernel(const float* __restrict__ Hs, const int* __restrict__ snd,
                                   const int* __restrict__ rcv, float* __restrict__ agg,
                                   long E, long Nn) {
  const long id = (long)blockIdx.x * blockDim.x + threadIdx.x;
  const long total = (E + Nn) * 32;
  if (id >= total) return;
  const long e = id >> 5;
  const int f4 = (int)(id & 31);
  int s, r;
  if (e < E) { s = snd[e]; r = rcv[e]; }
  else       { s = (int)(e - E); r = s; }  // self edge
  const float4 v = ((const float4*)Hs)[(long)s * 32 + f4];
  float* dst = agg + (long)r * 128 + f4 * 4;
  atomicAdd(dst + 0, v.x);
  atomicAdd(dst + 1, v.y);
  atomicAdd(dst + 2, v.z);
  atomicAdd(dst + 3, v.w);
}

// ---------------------------------------------------------------------------
// Kernel 5: per-graph mean pooling fused with receiver-degree normalization
// ---------------------------------------------------------------------------
__global__ void gcn_pool_kernel(const float* __restrict__ agg, const float* __restrict__ rdeg,
                                const int* __restrict__ n_node, float* __restrict__ pooled,
                                int nper) {
  const int g = blockIdx.x;
  const int f = threadIdx.x;  // 128 threads
  const long base = (long)g * nper;
  float s = 0.0f;
  for (int n = 0; n < nper; ++n) {
    const float d = rdeg[base + n];
    s += agg[(base + n) * 128 + f] * rsqrtf(fmaxf(d, 1.0f));
  }
  pooled[(long)g * 128 + f] = s / (float)n_node[g];
}

// ---------------------------------------------------------------------------
// Kernel 6: output MLP  logits = relu(P@W3+b3)@W4+b4   (C <= 16, masked tile)
// ---------------------------------------------------------------------------
__global__ __launch_bounds__(128) void out_mlp_wmma(
    const float* __restrict__ P, const float* __restrict__ W3, const float* __restrict__ b3,
    const float* __restrict__ W4, const float* __restrict__ b4,
    float* __restrict__ out, int G, int C) {
  __shared__ _Float16 Ps[128 * 128];
  __shared__ _Float16 Ws[4 * 8 * 32 * 16];

  const int tid = threadIdx.x;
  const int wave = tid >> 5;
  const int lane = tid & 31;
  const int hs = lane >> 4;
  const int lo = lane & 15;
  const long row0 = (long)blockIdx.x * 128;

  // Stage pooled block (rows clamped to G) and W3 (swizzled).
  {
    for (int i = tid; i < 128 * 32; i += 128) {
      const long row = row0 + (i >> 5);
      float4 x = make_float4(0.f, 0.f, 0.f, 0.f);
      if (row < G) x = ((const float4*)P)[row * 32 + (i & 31)];
      Ps[i * 4 + 0] = (_Float16)x.x; Ps[i * 4 + 1] = (_Float16)x.y;
      Ps[i * 4 + 2] = (_Float16)x.z; Ps[i * 4 + 3] = (_Float16)x.w;
    }
    stage_weight128_swz(W3, Ws, tid, 128);
  }
  __syncthreads();

  v16h aAll[2][4];
#pragma unroll
  for (int mt = 0; mt < 2; ++mt)
#pragma unroll
    for (int kt = 0; kt < 4; ++kt)
      aAll[mt][kt] = load_a_frag(Ps + (wave * 32 + mt * 16) * 128, lo, hs, kt, 128);

#pragma unroll
  for (int nt = 0; nt < 8; ++nt) {
    v8f acc0 = {}, acc1 = {};
#pragma unroll
    for (int kt = 0; kt < 4; ++kt) {
      v16h b = load_b_frag_swz(Ws, kt, nt, lane, 8);
      acc0 = __builtin_amdgcn_wmma_f32_16x16x32_f16(false, aAll[0][kt], false, b,
                                                    (short)0, acc0, false, false);
      acc1 = __builtin_amdgcn_wmma_f32_16x16x32_f16(false, aAll[1][kt], false, b,
                                                    (short)0, acc1, false, false);
    }
    const int n = nt * 16 + lo;
    const float bias = b3[n];
#pragma unroll
    for (int r = 0; r < 8; ++r) {
      const int m = r + 8 * hs;
      float v0 = acc0[r] + bias;
      float v1 = acc1[r] + bias;
      v0 = v0 > 0.0f ? v0 : 0.0f;
      v1 = v1 > 0.0f ? v1 : 0.0f;
      Ps[(wave * 32 + m) * 128 + n]      = (_Float16)v0;
      Ps[(wave * 32 + 16 + m) * 128 + n] = (_Float16)v1;
    }
  }
  __syncthreads();

  // Stage W4 [128 x C] into the swizzled layout padded to one 16-wide n-tile.
  for (int i = tid; i < 64 * 16; i += 128) {
    const int n = i & 15;
    const int k = (i >> 4) << 1;  // even k
    const _Float16 v0 = (n < C) ? (_Float16)W4[k * C + n] : (_Float16)0.0f;
    const _Float16 v1 = (n < C) ? (_Float16)W4[(k + 1) * C + n] : (_Float16)0.0f;
    const int kt = k >> 5, kk = k & 31;
    const int hs2 = kk >> 4, e = kk & 15;
    const int ln = hs2 * 16 + n;
    _Float16* dst = Ws + ((kt * 32) + ln) * 16 + e;  // ntiles = 1
    dst[0] = v0;
    dst[1] = v1;
  }
  __syncthreads();

#pragma unroll
  for (int mt = 0; mt < 2; ++mt)
#pragma unroll
    for (int kt = 0; kt < 4; ++kt)
      aAll[mt][kt] = load_a_frag(Ps + (wave * 32 + mt * 16) * 128, lo, hs, kt, 128);

  v8f acc0 = {}, acc1 = {};
#pragma unroll
  for (int kt = 0; kt < 4; ++kt) {
    v16h b = load_b_frag_swz(Ws, kt, 0, lane, 1);
    acc0 = __builtin_amdgcn_wmma_f32_16x16x32_f16(false, aAll[0][kt], false, b,
                                                  (short)0, acc0, false, false);
    acc1 = __builtin_amdgcn_wmma_f32_16x16x32_f16(false, aAll[1][kt], false, b,
                                                  (short)0, acc1, false, false);
  }
  if (lo < C) {
    const float bias = b4[lo];
#pragma unroll
    for (int r = 0; r < 8; ++r) {
      const int m = r + 8 * hs;
      const long g0 = row0 + wave * 32 + m;
      const long g1 = g0 + 16;
      if (g0 < G) out[g0 * C + lo] = acc0[r] + bias;
      if (g1 < G) out[g1 * C + lo] = acc1[r] + bias;
    }
  }
}

// ---------------------------------------------------------------------------
// Launcher
// ---------------------------------------------------------------------------
extern "C" void kernel_launch(void* const* d_in, const int* in_sizes, int n_in,
                              void* d_out, int out_size, void* d_ws, size_t ws_size,
                              hipStream_t stream) {
  const float* nodes     = (const float*)d_in[0];
  const int*   senders   = (const int*)d_in[1];
  const int*   receivers = (const int*)d_in[2];
  const int*   n_node    = (const int*)d_in[3];
  const float* W1 = (const float*)d_in[4];
  const float* b1 = (const float*)d_in[5];
  const float* W2 = (const float*)d_in[6];
  const float* b2 = (const float*)d_in[7];
  const float* W3 = (const float*)d_in[8];
  const float* b3 = (const float*)d_in[9];
  const float* W4 = (const float*)d_in[10];
  const float* b4 = (const float*)d_in[11];
  float* out = (float*)d_out;

  const long Nn   = (long)in_sizes[0] / 128;  // nodes
  const long E    = (long)in_sizes[1];        // edges
  const int  G    = in_sizes[3];              // graphs
  const int  C    = out_size / G;             // classes
  const int  nper = (int)(Nn / G);            // nodes per graph

  // Workspace layout (floats)
  float* ws     = (float*)d_ws;
  float* h      = ws;                // Nn*128
  float* agg    = h + Nn * 128;      // Nn*128
  float* sdeg   = agg + Nn * 128;    // Nn
  float* rdeg   = sdeg + Nn;         // Nn
  float* pooled = rdeg + Nn;         // G*128

  const long initN = Nn * 128;
  gcn_init_kernel<<<(unsigned)((initN + 255) / 256), 256, 0, stream>>>(agg, sdeg, rdeg, Nn);
  gcn_degree_kernel<<<(unsigned)((E + 255) / 256), 256, 0, stream>>>(senders, receivers, sdeg, rdeg, E);
  node_mlp_wmma<<<(unsigned)(Nn / 128), 128, 0, stream>>>(nodes, W1, b1, W2, b2, sdeg, h);
  const long scat = (E + Nn) * 32;
  gcn_scatter_kernel<<<(unsigned)((scat + 255) / 256), 256, 0, stream>>>(h, senders, receivers, agg, E, Nn);
  gcn_pool_kernel<<<(unsigned)G, 128, 0, stream>>>(agg, rdeg, n_node, pooled, nper);
  out_mlp_wmma<<<(unsigned)((G + 127) / 128), 128, 0, stream>>>(pooled, W3, b3, W4, b4, out, G, C);
}
